// DWFormerBlock_32066225832468
// MI455X (gfx1250) — matrix operand order repaired and checked by reading, not compiled, not run on hardware
//
#include <hip/hip_runtime.h>

// ---------------------------------------------------------------------------
// Types for CDNA5 WMMA (wave32, 16x16x32 bf16 -> f32)
// ---------------------------------------------------------------------------
typedef __attribute__((ext_vector_type(16))) __bf16 v16bf;
typedef __attribute__((ext_vector_type(8)))  __bf16 v8bf;
typedef __attribute__((ext_vector_type(8)))  float  v8f;

#define NEGBIG (-1.0e9f)

__device__ __forceinline__ __bf16 f2bf(float f) {
    unsigned u = __builtin_bit_cast(unsigned, f);
    unsigned r = u + 0x7FFFu + ((u >> 16) & 1u);   // RNE
    unsigned short h = (unsigned short)(r >> 16);
    return __builtin_bit_cast(__bf16, h);
}

__device__ __forceinline__ v8f wmma_bf16(v16bf a, v16bf b, v8f c) {
    return __builtin_amdgcn_wmma_f32_16x16x32_bf16(
        /*neg_a=*/false, a, /*neg_b=*/false, b,
        /*c_mod=*/(short)0, c, /*reuse_a=*/false, /*reuse_b=*/false);
}

// A-matrix (16x32 bf16) fragment: lane holds row (lane&15);
// k = e + (e>=8 ? 8 : 0) + (lane>=16 ? 8 : 0)   (ISA 7.12.2)
__device__ __forceinline__ v16bf ld_fragA(const __bf16* tile, int ld, int lane) {
    const __bf16* p = tile + (lane & 15) * ld + (((lane >> 4) & 1) << 3);
    v8bf lo = *(const v8bf*)(p);        // k 0..7  or 8..15
    v8bf hi = *(const v8bf*)(p + 16);   // k 16..23 or 24..31
    v16bf o;
#pragma unroll
    for (int i = 0; i < 8; i++) { o[i] = lo[i]; o[i + 8] = hi[i]; }
    return o;
}

// B-matrix (32x16 bf16) fragment from n-major LDS storage (tile[n][k]):
// lane holds col (lane&15); k = e + (e>=8 ? 8 : 0) + (lane>=16 ? 8 : 0)
__device__ __forceinline__ v16bf ld_fragB(const __bf16* tile, int ld, int lane) {
    const __bf16* p = tile + (lane & 15) * ld + (((lane >> 4) & 1) << 3);
    v8bf lo = *(const v8bf*)(p);
    v8bf hi = *(const v8bf*)(p + 16);
    v16bf o;
#pragma unroll
    for (int i = 0; i < 8; i++) { o[i] = lo[i]; o[i + 8] = hi[i]; }
    return o;
}

__device__ __forceinline__ v8f v8f_zero() {
    v8f z = {0.f, 0.f, 0.f, 0.f, 0.f, 0.f, 0.f, 0.f};
    return z;
}

// ---------------------------------------------------------------------------
// Block reductions (256 threads = 8 waves, wave32)
// ---------------------------------------------------------------------------
__device__ __forceinline__ float block_sum256(float v, float* red) {
#pragma unroll
    for (int o = 16; o > 0; o >>= 1) v += __shfl_xor(v, o, 32);
    __syncthreads();
    if ((threadIdx.x & 31) == 0) red[threadIdx.x >> 5] = v;
    __syncthreads();
    float r = red[0];
#pragma unroll
    for (int i = 1; i < 8; i++) r += red[i];
    return r;
}

__device__ __forceinline__ float block_max256(float v, float* red) {
#pragma unroll
    for (int o = 16; o > 0; o >>= 1) v = fmaxf(v, __shfl_xor(v, o, 32));
    __syncthreads();
    if ((threadIdx.x & 31) == 0) red[threadIdx.x >> 5] = v;
    __syncthreads();
    float r = red[0];
#pragma unroll
    for (int i = 1; i < 8; i++) r = fmaxf(r, red[i]);
    return r;
}

// ---------------------------------------------------------------------------
// Elementwise helpers
// ---------------------------------------------------------------------------
__global__ void conv_kernel(const float* __restrict__ in, __bf16* __restrict__ out, int n) {
    int i = blockIdx.x * 256 + threadIdx.x;
    if (i < n) out[i] = f2bf(in[i]);
}

__global__ void scale_kernel(const float* __restrict__ in, float* __restrict__ out,
                             float s, int n) {
    int i = blockIdx.x * 256 + threadIdx.x;
    if (i < n) out[i] = in[i] * s;
}

// ---------------------------------------------------------------------------
// Window / mask construction: one block per batch (T = 768)
// ---------------------------------------------------------------------------
__global__ __launch_bounds__(256) void masks_kernel(
    const float* __restrict__ halting, const float* __restrict__ maskin,
    int* __restrict__ wid, int* __restrict__ nbv,
    float* __restrict__ wise, float* __restrict__ thr_out) {
    const int T = 768;
    __shared__ float h[768];
    __shared__ int a1[768], a3[768], a3f[768], wid_s[768];
    __shared__ float red[8];
    __shared__ float sh_thr;
    const int b = blockIdx.x, tid = threadIdx.x;

    float acc = 0.f;
    for (int t = tid; t < T; t += 256) {
        h[t] = halting[b * T + t];
        acc += 1.f - maskin[b * T + t];
    }
    float m1 = block_sum256(acc, red);
    int med = (int)(m1 * 0.5f + (float)T - m1);
    med = med < 0 ? 0 : (med > T - 1 ? T - 1 : med);
    __syncthreads();

    // k-th order statistic via stable ranking (thr = sort(h)[med])
    for (int i = tid; i < T; i += 256) {
        float hi = h[i];
        int rank = 0;
        for (int j = 0; j < T; j++) {
            float hj = h[j];
            rank += (hj < hi) || (hj == hi && j < i);
        }
        if (rank == med) sh_thr = hi;
    }
    __syncthreads();
    const float thr = sh_thr;

    for (int t = tid; t < T; t += 256) a1[t] = (h[t] >= thr) ? 1 : 0;
    __syncthreads();
    for (int t = tid; t < T; t += 256) {
        int x1 = a1[t];
        int x2 = (t < T - 1) ? a1[t + 1] : (1 - x1);
        a3[t] = (x1 != x2) ? 1 : 0;          // x2-x1 with -1 mapped to 1
    }
    __syncthreads();
    for (int t = tid; t < T; t += 256) {
        int v = a3[t];
        if (t > 0 && v == 1 && a3[t - 1] == 1) v = 0;   // x4==2 -> 0
        if (t == T - 1) v = 1;
        a3f[t] = v;
    }
    __syncthreads();
    if (tid == 0) {
        int cs = 0;
        for (int t = 0; t < T; t++) { cs += a3f[t]; wid_s[t] = cs - a3f[t]; }
        nbv[b] = cs;
        thr_out[b] = thr;
    }
    __syncthreads();
    for (int t = tid; t < T; t += 256) {
        wid[b * T + t] = wid_s[t];
        wise[b * T + t] = a1[t] ? 1.0f : 0.85f;
    }
}

// ---------------------------------------------------------------------------
// LayerNorm over F=512, one block per row; optional bf16 copy of the output
// ---------------------------------------------------------------------------
__global__ __launch_bounds__(256) void ln_kernel(
    const float* __restrict__ in, float* __restrict__ out32,
    __bf16* __restrict__ out16) {
    __shared__ float red[8];
    const size_t base = (size_t)blockIdx.x * 512;
    const int t = threadIdx.x;
    float a0 = in[base + t], a1 = in[base + t + 256];
    float mean = block_sum256(a0 + a1, red) * (1.f / 512.f);
    float d0 = a0 - mean, d1 = a1 - mean;
    float var = block_sum256(d0 * d0 + d1 * d1, red) * (1.f / 512.f);
    float inv = rsqrtf(var + 1e-5f);
    float r0 = d0 * inv, r1 = d1 * inv;
    out32[base + t] = r0;
    out32[base + t + 256] = r1;
    if (out16) {
        out16[base + t] = f2bf(r0);
        out16[base + t + 256] = f2bf(r1);
    }
}

// ---------------------------------------------------------------------------
// Tiled bf16 WMMA GEMM: out[M,N] = A[M,K] @ B[K,N] (+resid, relu), f32 accum.
// Block tile 128x128x32, 8 waves (2x4), each wave 64x32 via 4x2 WMMA tiles.
// Software-pipelined with double-buffered LDS (ping-pong, 1 barrier/iter).
// Epilogue configuration is compile-time to keep the hot path branch-free.
// ---------------------------------------------------------------------------
template <bool RESID, bool RELU, bool OUT32, bool OUT16>
__global__ __launch_bounds__(256) void gemm_kernel(
    const __bf16* __restrict__ A, const __bf16* __restrict__ Bw,
    const float* __restrict__ resid, float* __restrict__ out32,
    __bf16* __restrict__ out16, int M, int N, int K) {
    __shared__ __bf16 As[2][128][40];   // 80B rows (16B aligned)
    __shared__ __bf16 Bs[2][128][40];   // n-major (transposed) B tile
    const int tid = threadIdx.x, lane = tid & 31, wave = tid >> 5;
    const int wm = wave >> 2, wn = wave & 3;
    const int m0 = blockIdx.y * 128, n0 = blockIdx.x * 128;

    // per-thread global load coordinates (fixed across tiles)
    const int ar0 = tid >> 2,        ako0 = (tid & 3) << 3;          // A chunk 0
    const int ar1 = (tid + 256) >> 2, ako1 = ((tid + 256) & 3) << 3; // A chunk 1
    const int bk0 = tid >> 4,        bno0 = (tid & 15) << 3;         // B chunk 0
    const int bk1 = (tid + 256) >> 4, bno1 = ((tid + 256) & 15) << 3;// B chunk 1

    v8f acc[4][2];
#pragma unroll
    for (int i = 0; i < 4; i++)
#pragma unroll
        for (int j = 0; j < 2; j++) acc[i][j] = v8f_zero();

    const int ntiles = K >> 5;

    // preload tile 0 into LDS buffer 0
    {
        v8bf a0 = *(const v8bf*)(A + (size_t)(m0 + ar0) * K + ako0);
        v8bf a1 = *(const v8bf*)(A + (size_t)(m0 + ar1) * K + ako1);
        v8bf b0 = *(const v8bf*)(Bw + (size_t)bk0 * N + (n0 + bno0));
        v8bf b1 = *(const v8bf*)(Bw + (size_t)bk1 * N + (n0 + bno1));
        *(v8bf*)&As[0][ar0][ako0] = a0;
        *(v8bf*)&As[0][ar1][ako1] = a1;
#pragma unroll
        for (int j = 0; j < 8; j++) Bs[0][bno0 + j][bk0] = b0[j];
#pragma unroll
        for (int j = 0; j < 8; j++) Bs[0][bno1 + j][bk1] = b1[j];
    }

    int cur = 0;
    for (int t = 0; t < ntiles; t++) {
        __syncthreads();
        // issue global loads for tile t+1 before computing tile t
        v8bf na0, na1, nb0, nb1;
        const bool more = (t + 1) < ntiles;
        if (more) {
            const int k0 = (t + 1) << 5;
            na0 = *(const v8bf*)(A + (size_t)(m0 + ar0) * K + (k0 + ako0));
            na1 = *(const v8bf*)(A + (size_t)(m0 + ar1) * K + (k0 + ako1));
            nb0 = *(const v8bf*)(Bw + (size_t)(k0 + bk0) * N + (n0 + bno0));
            nb1 = *(const v8bf*)(Bw + (size_t)(k0 + bk1) * N + (n0 + bno1));
            if (t + 2 < ntiles) {   // speculative L2 prefetch two tiles ahead
                __builtin_prefetch(A + (size_t)(m0 + ar0) * K + ((t + 2) << 5), 0, 1);
                __builtin_prefetch(Bw + (size_t)(((t + 2) << 5) + bk0) * N + n0, 0, 1);
            }
        }

        // compute tile t from LDS[cur]
        v16bf af[4], bfg[2];
#pragma unroll
        for (int i = 0; i < 4; i++) af[i] = ld_fragA(&As[cur][wm * 64 + i * 16][0], 40, lane);
#pragma unroll
        for (int j = 0; j < 2; j++) bfg[j] = ld_fragB(&Bs[cur][wn * 32 + j * 16][0], 40, lane);
#pragma unroll
        for (int i = 0; i < 4; i++)
#pragma unroll
            for (int j = 0; j < 2; j++) acc[i][j] = wmma_bf16(af[i], bfg[j], acc[i][j]);

        // stage tile t+1 into the other LDS buffer
        if (more) {
            const int nxt = cur ^ 1;
            *(v8bf*)&As[nxt][ar0][ako0] = na0;
            *(v8bf*)&As[nxt][ar1][ako1] = na1;
#pragma unroll
            for (int j = 0; j < 8; j++) Bs[nxt][bno0 + j][bk0] = nb0[j];
#pragma unroll
            for (int j = 0; j < 8; j++) Bs[nxt][bno1 + j][bk1] = nb1[j];
        }
        cur ^= 1;
    }

    // ----------------------- epilogue (branch-free) ------------------------
    const int half = (lane >> 4) & 1, nl = lane & 15;
#pragma unroll
    for (int i = 0; i < 4; i++) {
        const int row0 = m0 + wm * 64 + i * 16 + half * 8;
#pragma unroll
        for (int j = 0; j < 2; j++) {
            const int col = n0 + wn * 32 + j * 16 + nl;
            const size_t base = (size_t)row0 * N + col;
#pragma unroll
            for (int r = 0; r < 8; r++) {
                float v = acc[i][j][r];
                if constexpr (RESID) v += resid[base + (size_t)r * N];
                if constexpr (RELU) v = fmaxf(v, 0.f);
                if constexpr (OUT32) out32[base + (size_t)r * N] = v;
                if constexpr (OUT16) out16[base + (size_t)r * N] = f2bf(v);
            }
        }
    }
}

// ---------------------------------------------------------------------------
// Fused attention, per (b, h, 64-query tile). Two-pass exact softmax.
// MODE 0: local (wid[q]==wid[k] allowed); MODE 1: global (k < nb allowed).
// Also accumulates column sums of A into colsum (for A.mean(axis=(1,2))).
// ---------------------------------------------------------------------------
template <int MODE>
__global__ __launch_bounds__(256) void attn_kernel(
    const __bf16* __restrict__ Qg, const __bf16* __restrict__ Kg,
    const __bf16* __restrict__ Vg, const int* __restrict__ wid,
    const int* __restrict__ nbv, __bf16* __restrict__ Og,
    float* __restrict__ colsum) {
    const int T = 768, F = 512;
    const int b = blockIdx.z, h = blockIdx.y, qt = blockIdx.x;
    const int tid = threadIdx.x, lane = tid & 31, wave = tid >> 5;
    const int qw = wave >> 2, kw = wave & 3;
    const int half = (lane >> 4) & 1, nl = lane & 15;

    __shared__ __bf16 Qs[64][64];
    __shared__ __bf16 Ks[128][64];
    __shared__ __bf16 Vs[64][128];   // d-major (transposed V tile)
    __shared__ __bf16 Ps[64][128];   // normalized probabilities (bf16)
    __shared__ float rowM[64], rowL[64];
    __shared__ float stM[64][4], stL[64][4];
    __shared__ int widQ[64];

    const int nb = nbv[b];
    const size_t hb = ((size_t)b * T) * F + (size_t)h * 64;
    const float scale = 0.125f;   // 1/sqrt(64)

#pragma unroll
    for (int i = 0; i < 2; i++) {            // Q tile 64x64
        int idx = tid + i * 256;
        int r = idx >> 3, dd = (idx & 7) << 3;
        *(v8bf*)&Qs[r][dd] = *(const v8bf*)(Qg + hb + (size_t)(qt * 64 + r) * F + dd);
    }
    if (tid < 64) {
        widQ[tid] = wid[b * T + qt * 64 + tid];
        rowM[tid] = -3.0e38f;
        rowL[tid] = 0.f;
    }

    // ------------------------- PASS 1: row max / sum-exp -------------------
    for (int kt = 0; kt < T; kt += 128) {
        __syncthreads();
#pragma unroll
        for (int i = 0; i < 4; i++) {        // K tile 128x64
            int idx = tid + i * 256;
            int r = idx >> 3, dd = (idx & 7) << 3;
            *(v8bf*)&Ks[r][dd] = *(const v8bf*)(Kg + hb + (size_t)(kt + r) * F + dd);
        }
        __syncthreads();

        v8f s[2][2];
#pragma unroll
        for (int i = 0; i < 2; i++)
#pragma unroll
            for (int j = 0; j < 2; j++) s[i][j] = v8f_zero();
#pragma unroll
        for (int c = 0; c < 2; c++) {
            v16bf a0 = ld_fragA(&Qs[qw * 32][c * 32], 64, lane);
            v16bf a1 = ld_fragA(&Qs[qw * 32 + 16][c * 32], 64, lane);
            v16bf b0 = ld_fragB(&Ks[kw * 32][c * 32], 64, lane);
            v16bf b1 = ld_fragB(&Ks[kw * 32 + 16][c * 32], 64, lane);
            s[0][0] = wmma_bf16(a0, b0, s[0][0]);
            s[0][1] = wmma_bf16(a0, b1, s[0][1]);
            s[1][0] = wmma_bf16(a1, b0, s[1][0]);
            s[1][1] = wmma_bf16(a1, b1, s[1][1]);
        }

        const int k0g = kt + kw * 32 + nl;
        int wk0 = 0, wk1 = 0;
        bool ok0 = true, ok1 = true;
        if constexpr (MODE == 0) {
            wk0 = wid[b * T + k0g];
            wk1 = wid[b * T + k0g + 16];
        } else {
            ok0 = (k0g < nb);
            ok1 = (k0g + 16 < nb);
        }

#pragma unroll
        for (int i = 0; i < 2; i++)
#pragma unroll
            for (int r = 0; r < 8; r++) {
                int qloc = qw * 32 + i * 16 + r + half * 8;
                float v0 = s[i][0][r] * scale, v1 = s[i][1][r] * scale;
                if constexpr (MODE == 0) {
                    int wq = widQ[qloc];
                    if (wq != wk0) v0 = NEGBIG;
                    if (wq != wk1) v1 = NEGBIG;
                } else {
                    if (!ok0) v0 = NEGBIG;
                    if (!ok1) v1 = NEGBIG;
                }
                float mx = fmaxf(v0, v1);
#pragma unroll
                for (int o = 1; o < 16; o <<= 1) mx = fmaxf(mx, __shfl_xor(mx, o, 32));
                float e = __expf(v0 - mx) + __expf(v1 - mx);
#pragma unroll
                for (int o = 1; o < 16; o <<= 1) e += __shfl_xor(e, o, 32);
                if (nl == 0) { stM[qloc][kw] = mx; stL[qloc][kw] = e; }
            }
        __syncthreads();
        if (tid < 64) {    // online merge across the 4 k-waves
            float m = rowM[tid], l = rowL[tid];
#pragma unroll
            for (int w2 = 0; w2 < 4; w2++) {
                float mw = stM[tid][w2], lw = stL[tid][w2];
                float mn = fmaxf(m, mw);
                l = l * __expf(m - mn) + lw * __expf(mw - mn);
                m = mn;
            }
            rowM[tid] = m;
            rowL[tid] = l;
        }
    }
    __syncthreads();

    // ------------------------- PASS 2: O = P @ V, column sums --------------
    v8f o0 = v8f_zero(), o1 = v8f_zero();
    for (int kt = 0; kt < T; kt += 128) {
        __syncthreads();
#pragma unroll
        for (int i = 0; i < 4; i++) {        // K tile + transposed V tile
            int idx = tid + i * 256;
            int r = idx >> 3, dd = (idx & 7) << 3;
            *(v8bf*)&Ks[r][dd] = *(const v8bf*)(Kg + hb + (size_t)(kt + r) * F + dd);
            v8bf vv = *(const v8bf*)(Vg + hb + (size_t)(kt + r) * F + dd);
#pragma unroll
            for (int j = 0; j < 8; j++) Vs[dd + j][r] = vv[j];
        }
        __syncthreads();

        v8f s[2][2];
#pragma unroll
        for (int i = 0; i < 2; i++)
#pragma unroll
            for (int j = 0; j < 2; j++) s[i][j] = v8f_zero();
#pragma unroll
        for (int c = 0; c < 2; c++) {
            v16bf a0 = ld_fragA(&Qs[qw * 32][c * 32], 64, lane);
            v16bf a1 = ld_fragA(&Qs[qw * 32 + 16][c * 32], 64, lane);
            v16bf b0 = ld_fragB(&Ks[kw * 32][c * 32], 64, lane);
            v16bf b1 = ld_fragB(&Ks[kw * 32 + 16][c * 32], 64, lane);
            s[0][0] = wmma_bf16(a0, b0, s[0][0]);
            s[0][1] = wmma_bf16(a0, b1, s[0][1]);
            s[1][0] = wmma_bf16(a1, b0, s[1][0]);
            s[1][1] = wmma_bf16(a1, b1, s[1][1]);
        }

        const int k0g = kt + kw * 32 + nl;
        int wk0 = 0, wk1 = 0;
        bool ok0 = true, ok1 = true;
        if constexpr (MODE == 0) {
            wk0 = wid[b * T + k0g];
            wk1 = wid[b * T + k0g + 16];
        } else {
            ok0 = (k0g < nb);
            ok1 = (k0g + 16 < nb);
        }

        float cs0 = 0.f, cs1 = 0.f;
#pragma unroll
        for (int i = 0; i < 2; i++)
#pragma unroll
            for (int r = 0; r < 8; r++) {
                int qloc = qw * 32 + i * 16 + r + half * 8;
                float v0 = s[i][0][r] * scale, v1 = s[i][1][r] * scale;
                if constexpr (MODE == 0) {
                    int wq = widQ[qloc];
                    if (wq != wk0) v0 = NEGBIG;
                    if (wq != wk1) v1 = NEGBIG;
                } else {
                    if (!ok0) v0 = NEGBIG;
                    if (!ok1) v1 = NEGBIG;
                }
                float rm = rowM[qloc];
                float inv = 1.0f / rowL[qloc];
                float p0 = __expf(v0 - rm) * inv;
                float p1 = __expf(v1 - rm) * inv;
                Ps[qloc][kw * 32 + nl] = f2bf(p0);
                Ps[qloc][kw * 32 + 16 + nl] = f2bf(p1);
                cs0 += p0;
                cs1 += p1;
            }
        cs0 += __shfl_xor(cs0, 16, 32);
        cs1 += __shfl_xor(cs1, 16, 32);
        if (lane < 16) {
            atomicAdd(&colsum[b * T + kt + kw * 32 + lane], cs0);
            atomicAdd(&colsum[b * T + kt + kw * 32 + 16 + lane], cs1);
        }
        __syncthreads();

        // phase B: waves re-mapped as (qw, dw=kw) -> O[32q x 16d]
#pragma unroll
        for (int kc = 0; kc < 4; kc++) {
            v16bf pa0 = ld_fragA(&Ps[qw * 32][kc * 32], 128, lane);
            v16bf pa1 = ld_fragA(&Ps[qw * 32 + 16][kc * 32], 128, lane);
            v16bf vb = ld_fragB(&Vs[kw * 16][kc * 32], 128, lane);
            o0 = wmma_bf16(pa0, vb, o0);
            o1 = wmma_bf16(pa1, vb, o1);
        }
    }

#pragma unroll
    for (int r = 0; r < 8; r++) {
        int q0 = qt * 64 + qw * 32 + r + half * 8;
        int dcol = kw * 16 + nl;
        Og[hb + (size_t)q0 * F + dcol] = f2bf(o0[r]);
        Og[hb + (size_t)(q0 + 16) * F + dcol] = f2bf(o1[r]);
    }
}

// ---------------------------------------------------------------------------
// local_x = y * wise (in place); scatter local_x*local_score into pre windows
// ---------------------------------------------------------------------------
__global__ __launch_bounds__(256) void finalize_local_kernel(
    float* __restrict__ lx, const float* __restrict__ wise,
    const float* __restrict__ ls, const int* __restrict__ wid,
    float* __restrict__ pre) {
    const int T = 768, F = 512;
    const int bt = blockIdx.x, b = bt / T;
    const float w = wise[bt], s = ls[bt];
    const int wd = wid[bt];
    const size_t src = (size_t)bt * F;
    const size_t dst = ((size_t)b * T + wd) * F;
    for (int f = threadIdx.x; f < F; f += 256) {
        float v = lx[src + f] * w;
        lx[src + f] = v;
        atomicAdd(&pre[dst + f], v * s);
    }
}

// data = local_x + gather(global_x, wid)
__global__ __launch_bounds__(256) void final_data_kernel(
    const float* __restrict__ lx, const float* __restrict__ gx,
    const int* __restrict__ wid, float* __restrict__ out) {
    const int T = 768, F = 512;
    const int bt = blockIdx.x, b = bt / T;
    const size_t src = (size_t)bt * F;
    const size_t g = ((size_t)b * T + wid[bt]) * F;
    for (int f = threadIdx.x; f < F; f += 256) out[src + f] = lx[src + f] + gx[g + f];
}

// attn = softmax(local_score * gather(hh2, wid)) per batch row
__global__ __launch_bounds__(256) void final_attn_kernel(
    const float* __restrict__ ls, const float* __restrict__ hh2,
    const int* __restrict__ wid, float* __restrict__ out) {
    const int T = 768;
    const int b = blockIdx.x, tid = threadIdx.x;
    __shared__ float z[768];
    __shared__ float red[8];
    for (int t = tid; t < T; t += 256)
        z[t] = ls[b * T + t] * hh2[b * T + wid[b * T + t]];
    __syncthreads();
    float mx = -3.0e38f;
    for (int t = tid; t < T; t += 256) mx = fmaxf(mx, z[t]);
    mx = block_max256(mx, red);
    float s = 0.f;
    for (int t = tid; t < T; t += 256) s += __expf(z[t] - mx);
    s = block_sum256(s, red);
    const float inv = 1.f / s;
    for (int t = tid; t < T; t += 256) out[b * T + t] = __expf(z[t] - mx) * inv;
}

// ---------------------------------------------------------------------------
// Launcher
// ---------------------------------------------------------------------------
extern "C" void kernel_launch(void* const* d_in, const int* in_sizes, int n_in,
                              void* d_out, int out_size, void* d_ws, size_t ws_size,
                              hipStream_t stream) {
    (void)in_sizes; (void)n_in; (void)out_size; (void)ws_size;
    const int B = 16, T = 768, F = 512, FF = 2048, H = 8;
    const size_t S1 = (size_t)B * T * F;
    const size_t S2 = (size_t)B * T * FF;
    const size_t BT = (size_t)B * T;

    char* p = (char*)d_ws;
    auto alloc = [&](size_t bytes) -> void* {
        void* q = (void*)p;
        p += (bytes + 255) & ~(size_t)255;
        return q;
    };

    const size_t wsz[12] = {(size_t)F * F, (size_t)F * F, (size_t)F * F, (size_t)F * F,
                            (size_t)F * FF, (size_t)FF * F,
                            (size_t)F * F, (size_t)F * F, (size_t)F * F, (size_t)F * F,
                            (size_t)F * FF, (size_t)FF * F};
    __bf16* wb[12];
    for (int i = 0; i < 12; i++) wb[i] = (__bf16*)alloc(wsz[i] * 2);

    int* wid    = (int*)alloc(BT * 4);
    int* nbv    = (int*)alloc(B * 4);
    float* wise = (float*)alloc(BT * 4);
    float* ls   = (float*)alloc(BT * 4);
    float* hh2  = (float*)alloc(BT * 4);
    float* csum = (float*)alloc(BT * 4);
    float* xl32 = (float*)alloc(S1 * 4);
    __bf16* xl16 = (__bf16*)alloc(S1 * 2);
    __bf16* q16 = (__bf16*)alloc(S1 * 2);
    __bf16* k16 = (__bf16*)alloc(S1 * 2);
    __bf16* v16 = (__bf16*)alloc(S1 * 2);
    __bf16* o16 = (__bf16*)alloc(S1 * 2);
    float* t32  = (float*)alloc(S1 * 4);
    float* y32  = (float*)alloc(S1 * 4);
    __bf16* y16 = (__bf16*)alloc(S1 * 2);
    __bf16* h16 = (__bf16*)alloc(S2 * 2);
    float* t32b = (float*)alloc(S1 * 4);
    float* lxw  = (float*)alloc(S1 * 4);
    float* gx   = (float*)alloc(S1 * 4);
    float* pre  = (float*)alloc(S1 * 4);
    __bf16* pre16 = (__bf16*)alloc(S1 * 2);

    float* out_data = (float*)d_out;
    float* out_thr  = out_data + S1;
    float* out_attn = out_thr + B;

    // weights f32 -> bf16
    for (int i = 0; i < 12; i++) {
        int n = (int)wsz[i];
        conv_kernel<<<(n + 255) / 256, 256, 0, stream>>>((const float*)d_in[3 + i], wb[i], n);
    }

    masks_kernel<<<B, 256, 0, stream>>>((const float*)d_in[1], (const float*)d_in[2],
                                        wid, nbv, wise, out_thr);
    ln_kernel<<<B * T, 256, 0, stream>>>((const float*)d_in[0], xl32, xl16);

    const dim3 blk(256);
    const dim3 g512(F / 128, (B * T) / 128);
    const dim3 g2048(FF / 128, (B * T) / 128);
    const dim3 ga(T / 64, H, B);
    const float inv_ht = 1.0f / (float)(H * T);

    // epilogue variants: <RESID, RELU, OUT32, OUT16>
    auto gemm_b16  = gemm_kernel<false, false, false, true>;   // plain -> bf16
    auto gemm_res  = gemm_kernel<true,  false, true,  false>;  // +resid -> f32
    auto gemm_relu = gemm_kernel<false, true,  false, true>;   // relu -> bf16

    // ---- local transformer ----
    gemm_b16<<<g512, blk, 0, stream>>>(xl16, wb[0], nullptr, nullptr, q16, B * T, F, F);
    gemm_b16<<<g512, blk, 0, stream>>>(xl16, wb[1], nullptr, nullptr, k16, B * T, F, F);
    gemm_b16<<<g512, blk, 0, stream>>>(xl16, wb[2], nullptr, nullptr, v16, B * T, F, F);
    hipMemsetAsync(csum, 0, BT * 4, stream);
    attn_kernel<0><<<ga, blk, 0, stream>>>(q16, k16, v16, wid, nbv, o16, csum);
    gemm_res<<<g512, blk, 0, stream>>>(o16, wb[3], xl32, t32, nullptr, B * T, F, F);
    ln_kernel<<<B * T, 256, 0, stream>>>(t32, y32, y16);
    gemm_relu<<<g2048, blk, 0, stream>>>(y16, wb[4], nullptr, nullptr, h16, B * T, FF, F);
    gemm_res<<<g512, blk, 0, stream>>>(h16, wb[5], y32, t32b, nullptr, B * T, F, FF);
    ln_kernel<<<B * T, 256, 0, stream>>>(t32b, lxw, nullptr);
    scale_kernel<<<((int)BT + 255) / 256, 256, 0, stream>>>(csum, ls, inv_ht, (int)BT);

    // ---- window pooling (pre = mm @ (local_x * local_score)) ----
    hipMemsetAsync(pre, 0, S1 * 4, stream);
    finalize_local_kernel<<<B * T, 256, 0, stream>>>(lxw, wise, ls, wid, pre);
    ln_kernel<<<B * T, 256, 0, stream>>>(pre, pre, pre16);   // in-place LN

    // ---- global transformer ----
    gemm_b16<<<g512, blk, 0, stream>>>(pre16, wb[6], nullptr, nullptr, q16, B * T, F, F);
    gemm_b16<<<g512, blk, 0, stream>>>(pre16, wb[7], nullptr, nullptr, k16, B * T, F, F);
    gemm_b16<<<g512, blk, 0, stream>>>(pre16, wb[8], nullptr, nullptr, v16, B * T, F, F);
    hipMemsetAsync(csum, 0, BT * 4, stream);
    attn_kernel<1><<<ga, blk, 0, stream>>>(q16, k16, v16, wid, nbv, o16, csum);
    gemm_res<<<g512, blk, 0, stream>>>(o16, wb[9], pre, t32, nullptr, B * T, F, F);
    ln_kernel<<<B * T, 256, 0, stream>>>(t32, y32, y16);
    gemm_relu<<<g2048, blk, 0, stream>>>(y16, wb[10], nullptr, nullptr, h16, B * T, FF, F);
    gemm_res<<<g512, blk, 0, stream>>>(h16, wb[11], y32, t32b, nullptr, B * T, F, FF);
    ln_kernel<<<B * T, 256, 0, stream>>>(t32b, gx, nullptr);
    scale_kernel<<<((int)BT + 255) / 256, 256, 0, stream>>>(csum, hh2, inv_ht, (int)BT);

    // ---- outputs ----
    final_data_kernel<<<B * T, 256, 0, stream>>>(lxw, gx, wid, out_data);
    final_attn_kernel<<<B, 256, 0, stream>>>(ls, hh2, wid, out_attn);
}
